// SH_INIT_37606733644281
// MI455X (gfx1250) — matrix operand order repaired and checked by reading, not compiled, not run on hardware
//
#include <hip/hip_runtime.h>
#include <hip/hip_bf16.h>
#include <math.h>

// ---------------------------------------------------------------------------
// EGNN edge-MLP + SH-scatter for MI455X (gfx1250, wave32, WMMA)
//   GEMM1: [128e x 288k] x [288k x 128n] f16 -> f32 via v_wmma_f32_16x16x32_f16
//   GEMM2: [128e x 128k] x [128k x 16n(pad 3)]            same instruction
// ---------------------------------------------------------------------------

typedef _Float16 h8   __attribute__((ext_vector_type(8)));
typedef _Float16 v16h __attribute__((ext_vector_type(16)));
typedef float    v8f  __attribute__((ext_vector_type(8)));

union H16 { v16h v; h8 h[2]; };

#define TILE_E 128
#define KPAD   288      // 273 padded to 9 chunks of 32
#define HID    128

// LDS layout (bytes), all 16B aligned
#define OFF_W1T   0                                   // 128*288*2 = 73728
#define OFF_MSG   (OFF_W1T + 128*KPAD*2)              // 73728
#define OFF_H     (OFF_MSG + TILE_E*KPAD*2)           // 128*128*2 = 32768
#define OFF_W2T   (OFF_H   + TILE_E*HID*2)            // 16*128*2  = 4096
#define OFF_DIFF  (OFF_W2T + 16*HID*2)                // 128*4*4   = 2048
#define OFF_IDX   (OFF_DIFF + TILE_E*4*4)             // 128*2*4   = 1024
#define OFF_WOUT  (OFF_IDX  + TILE_E*2*4)             // 128*3*4   = 1536
#define SMEM_BYTES (OFF_WOUT + TILE_E*3*4)            // 188928

__global__ void egnn_zero_ws(float* p, int n) {
    int i = blockIdx.x * blockDim.x + threadIdx.x;
    if (i < n) p[i] = 0.0f;
}

__device__ __forceinline__ float silu(float x) {
    return x / (1.0f + __expf(-x));
}

__global__ __launch_bounds__(256, 1)
void egnn_edge_kernel(const float* __restrict__ feat,
                      const float* __restrict__ pos,
                      const float* __restrict__ eattr,
                      const float* __restrict__ W1,
                      const float* __restrict__ b1,
                      const float* __restrict__ W2,
                      const float* __restrict__ b2,
                      const int*   __restrict__ eidx,
                      float* __restrict__ sum,
                      float* __restrict__ cnt,
                      int N, int E) {
    extern __shared__ char smem[];
    _Float16* w1t  = (_Float16*)(smem + OFF_W1T);   // [n][k] B-layout source
    _Float16* msg  = (_Float16*)(smem + OFF_MSG);   // [e][k]
    _Float16* hbuf = (_Float16*)(smem + OFF_H);     // [e][n]
    _Float16* w2t  = (_Float16*)(smem + OFF_W2T);   // [n(16)][k(128)]
    float*    dif  = (float*)   (smem + OFF_DIFF);  // [e][x,y,z,dist]
    int*      idxb = (int*)     (smem + OFF_IDX);   // [e][row,col]
    float*    wout = (float*)   (smem + OFF_WOUT);  // [e][3]

    const int tid   = threadIdx.x;
    const int wave  = tid >> 5;
    const int lane  = tid & 31;
    const int lrow  = lane & 15;     // M (A) or N (B/C) index within tile
    const int lhi   = lane >> 4;     // lane-half select
    const long tile0 = (long)blockIdx.x * TILE_E;

    // ---- Phase 0: edge metadata (diff, dist, indices, count atomic) -------
    if (tid < TILE_E) {
        int e = tid;
        long ge = tile0 + e;
        if (ge < E) {
            int r = eidx[ge];
            int c = eidx[(long)E + ge];
            float dx = pos[(long)r*3+0] - pos[(long)c*3+0];
            float dy = pos[(long)r*3+1] - pos[(long)c*3+1];
            float dz = pos[(long)r*3+2] - pos[(long)c*3+2];
            float d  = sqrtf(dx*dx + dy*dy + dz*dz);
            idxb[e*2+0] = r; idxb[e*2+1] = c;
            dif[e*4+0] = dx; dif[e*4+1] = dy; dif[e*4+2] = dz; dif[e*4+3] = d;
            unsafeAtomicAdd(cnt + r, 1.0f);
        } else {
            idxb[e*2+0] = -1; idxb[e*2+1] = 0;
            dif[e*4+0] = 0.f; dif[e*4+1] = 0.f; dif[e*4+2] = 0.f; dif[e*4+3] = 0.f;
        }
    }
    __syncthreads();

    // ---- Phase 1: stage W1^T (f16), W2 padded, msg tile (f16) -------------
    for (int i = tid; i < KPAD*128; i += blockDim.x) {     // coalesced over n
        int k = i >> 7, n = i & 127;
        float v = (k < 273) ? W1[(long)k*HID + n] : 0.0f;
        w1t[n*KPAD + k] = (_Float16)v;
    }
    for (int i = tid; i < 16*HID; i += blockDim.x) {
        int n = i >> 7, k = i & 127;
        float v = (n < 3) ? W2[(long)k*3 + n] : 0.0f;
        w2t[n*HID + k] = (_Float16)v;
    }
    for (int i = tid; i < TILE_E*KPAD; i += blockDim.x) {
        int e = i / KPAD, k = i - e*KPAD;
        int r = idxb[e*2+0];
        float v = 0.0f;
        if (r >= 0) {
            if (k == 0)          v = dif[e*4+3];
            else if (k < 129)    v = feat[(long)r*HID + (k-1)];
            else if (k < 257)    v = feat[(long)idxb[e*2+1]*HID + (k-129)];
            else if (k < 273)    v = eattr[(tile0 + e)*16 + (k-257)];
        }
        msg[e*KPAD + k] = (_Float16)v;
    }
    __syncthreads();

    // ---- Phase 2: GEMM1  h = silu(msg @ W1 + b1) --------------------------
    // wave owns edges [wave*16, wave*16+16), all 128 hidden cols (8 N-tiles)
    v8f acc[8];
#pragma unroll
    for (int nt = 0; nt < 8; ++nt) acc[nt] = (v8f){0,0,0,0,0,0,0,0};

#pragma unroll
    for (int kc = 0; kc < 9; ++kc) {
        // A: 16x32 f16, lanes 0-15: K 0-7 & 16-23 ; lanes 16-31: K 8-15 & 24-31
        const _Float16* ar = msg + (wave*16 + lrow)*KPAD + kc*32;
        H16 a;
        a.h[0] = *(const h8*)(ar + lhi*8);
        a.h[1] = *(const h8*)(ar + 16 + lhi*8);
#pragma unroll
        for (int nt = 0; nt < 8; ++nt) {
            // B: 32x16 f16, lanes 0-15: K 0-7 & 16-23 of col n ; lanes 16-31: K 8-15 & 24-31
            const _Float16* br = w1t + (nt*16 + lrow)*KPAD + kc*32 + lhi*8;
            H16 b;
            b.h[0] = *(const h8*)br;
            b.h[1] = *(const h8*)(br + 16);
            acc[nt] = __builtin_amdgcn_wmma_f32_16x16x32_f16(
                false, a.v, false, b.v, (short)0, acc[nt], false, false);
        }
    }

    // bias + silu, spill h (f16) to LDS in [e][n] order
#pragma unroll
    for (int nt = 0; nt < 8; ++nt) {
        int n = nt*16 + lrow;
        float bias = b1[n];
#pragma unroll
        for (int r = 0; r < 8; ++r) {
            int e = wave*16 + r + lhi*8;            // C layout: M = r (+8 hi half)
            hbuf[e*HID + n] = (_Float16)silu(acc[nt][r] + bias);
        }
    }
    __syncthreads();

    // ---- Phase 3: GEMM2  w = silu(h @ W2pad + b2), N padded to 16 ---------
    v8f acc2 = (v8f){0,0,0,0,0,0,0,0};
#pragma unroll
    for (int kc = 0; kc < 4; ++kc) {
        const _Float16* ar = hbuf + (wave*16 + lrow)*HID + kc*32;
        H16 a;
        a.h[0] = *(const h8*)(ar + lhi*8);
        a.h[1] = *(const h8*)(ar + 16 + lhi*8);
        const _Float16* br = w2t + lrow*HID + kc*32 + lhi*8;
        H16 b;
        b.h[0] = *(const h8*)br;
        b.h[1] = *(const h8*)(br + 16);
        acc2 = __builtin_amdgcn_wmma_f32_16x16x32_f16(
            false, a.v, false, b.v, (short)0, acc2, false, false);
    }
    if (lrow < 3) {
        float bias = b2[lrow];
#pragma unroll
        for (int r = 0; r < 8; ++r) {
            int e = wave*16 + r + lhi*8;
            wout[e*3 + lrow] = silu(acc2[r] + bias);
        }
    }
    __syncthreads();

    // ---- Phase 4: spherical harmonics * per-l weight, scatter-add ---------
    if (tid < TILE_E) {
        int e = tid;
        int r = idxb[e*2+0];
        if (r >= 0) {
            float dx = dif[e*4+0], dy = dif[e*4+1], dz = dif[e*4+2];
            float d  = dif[e*4+3];
            float inv = 1.0f / fmaxf(d, 1e-12f);
            float x = dx*inv, y = dy*inv, z = dz*inv;
            const float s3 = 1.7320508075688772f;
            float sh[9] = { 1.0f, x, y, z,
                            s3*x*z, s3*x*y,
                            y*y - 0.5f*(x*x + z*z),
                            s3*y*z, 0.5f*s3*(z*z - x*x) };
            float w0 = wout[e*3+0], w1v = wout[e*3+1], w2v = wout[e*3+2];
            float wl[9] = { w0, w1v, w1v, w1v, w2v, w2v, w2v, w2v, w2v };
            float* base = sum + (long)r*9;
#pragma unroll
            for (int c = 0; c < 9; ++c)
                unsafeAtomicAdd(base + c, sh[c] * wl[c]);
        }
    }
}

__global__ void egnn_finalize(const float* __restrict__ sum,
                              const float* __restrict__ cnt,
                              float* __restrict__ out, int N) {
    int i = blockIdx.x * blockDim.x + threadIdx.x;
    if (i < N*9) {
        int n = i / 9;
        out[i] = sum[i] / fmaxf(cnt[n], 1.0f);
    }
}

extern "C" void kernel_launch(void* const* d_in, const int* in_sizes, int n_in,
                              void* d_out, int out_size, void* d_ws, size_t ws_size,
                              hipStream_t stream) {
    const float* feat  = (const float*)d_in[0];
    const float* pos   = (const float*)d_in[1];
    const float* eattr = (const float*)d_in[2];
    const float* W1    = (const float*)d_in[3];
    const float* b1    = (const float*)d_in[4];
    const float* W2    = (const float*)d_in[5];
    const float* b2    = (const float*)d_in[6];
    const int*   eidx  = (const int*)d_in[7];

    const int N = in_sizes[0] / HID;       // 50000
    const int E = in_sizes[7] / 2;         // 1000000

    float* sum = (float*)d_ws;             // N*9
    float* cnt = sum + (size_t)N * 9;      // N

    int zn = N * 10;
    egnn_zero_ws<<<(zn + 255) / 256, 256, 0, stream>>>(sum, zn);

    int tiles = (E + TILE_E - 1) / TILE_E;
    egnn_edge_kernel<<<tiles, 256, SMEM_BYTES, stream>>>(
        feat, pos, eattr, W1, b1, W2, b2, eidx, sum, cnt, N, E);

    egnn_finalize<<<(N * 9 + 255) / 256, 256, 0, stream>>>(sum, cnt, (float*)d_out, N);
}
